// GCN_Fusion_70669391888404
// MI455X (gfx1250) — compile-verified, matrix operand order checked by reading, not run on hardware
//
#include <hip/hip_runtime.h>
#include <hip/hip_bf16.h>

#define B_ 2
#define V_ 1024
#define DIM_ 128
#define ROWS_ (B_ * V_)   // 2048
#define KMAX_ 100

typedef __attribute__((ext_vector_type(16))) __bf16 v16bf;
typedef __attribute__((ext_vector_type(2)))  __bf16 v2bf;
typedef __attribute__((ext_vector_type(8)))  float  v8f;

// ---------------------------------------------------------------------------
// kNN: one block per (b,v). Stage batch positions + dist in LDS, then 101
// rounds of block-wide argmin (self is selected first and skipped).
// ---------------------------------------------------------------------------
__global__ __launch_bounds__(128) void knn_kernel(const float* __restrict__ verts,
                                                  int* __restrict__ idx) {
  const int bv = blockIdx.x;
  const int b  = bv / V_;
  const int v  = bv % V_;
  __shared__ float spos[V_ * 3];
  __shared__ float sdist[V_];
  __shared__ float rval[128];
  __shared__ int   ridx[128];

  const float* vb = verts + (size_t)b * V_ * 3;
  for (int t = threadIdx.x; t < V_ * 3; t += 128) spos[t] = vb[t];
  __syncthreads();

  const float px = spos[v * 3 + 0], py = spos[v * 3 + 1], pz = spos[v * 3 + 2];
  for (int j = threadIdx.x; j < V_; j += 128) {
    float dx = spos[j * 3 + 0] - px;
    float dy = spos[j * 3 + 1] - py;
    float dz = spos[j * 3 + 2] - pz;
    sdist[j] = dx * dx + dy * dy + dz * dz;
  }
  __syncthreads();

  for (int s = 0; s <= KMAX_; ++s) {
    float best = 3.0e38f; int bi = 0;
    for (int j = threadIdx.x; j < V_; j += 128) {
      float d = sdist[j];
      if (d < best) { best = d; bi = j; }
    }
    rval[threadIdx.x] = best; ridx[threadIdx.x] = bi;
    __syncthreads();
    for (int w = 64; w > 0; w >>= 1) {
      if (threadIdx.x < w && rval[threadIdx.x + w] < rval[threadIdx.x]) {
        rval[threadIdx.x] = rval[threadIdx.x + w];
        ridx[threadIdx.x] = ridx[threadIdx.x + w];
      }
      __syncthreads();
    }
    if (threadIdx.x == 0) {
      int sel = ridx[0];
      if (s > 0) idx[(size_t)bv * KMAX_ + (s - 1)] = sel;  // skip self
      sdist[sel] = 3.0e38f;
    }
    __syncthreads();
  }
}

// ---------------------------------------------------------------------------
// Normalize direction columns: sd[:,k] = dirs[:,k] / max(||dirs[:,k]||, eps)
// ---------------------------------------------------------------------------
__global__ void sd_kernel(const float* __restrict__ dirs, float* __restrict__ sd) {
  const int k = threadIdx.x;  // 128 threads
  float a = dirs[k], b = dirs[DIM_ + k], c = dirs[2 * DIM_ + k];
  float n = sqrtf(a * a + b * b + c * c);
  n = fmaxf(n, 1e-12f);
  sd[k] = a / n; sd[DIM_ + k] = b / n; sd[2 * DIM_ + k] = c / n;
}

// ---------------------------------------------------------------------------
// Pre-swizzle fp32 weights (K x N row-major) into bf16 WMMA B-fragment order:
//   Wsw[ ((kt*(N/16) + ntile)*32 + lane)*16 + e ]
// so each lane's 16-element fragment is one contiguous 32-byte load.
// Fragment map (16-bit B 32x16, wave32): lane = n%16 + 16*(g&1),
// e = kk%8 + 8*(g>>1), g = kk/8, kk = k%32.
// ---------------------------------------------------------------------------
__global__ void wswizzle_kernel(const float* __restrict__ W, __bf16* __restrict__ Wsw,
                                int K, int N, int total) {
  const int i = blockIdx.x * blockDim.x + threadIdx.x;
  if (i >= total) return;
  const int k = i / N, n = i % N;
  const int kt = k >> 5, kk = k & 31, g = kk >> 3;
  const int lane = (n & 15) + ((g & 1) << 4);
  const int e    = (kk & 7) + ((g >> 1) << 3);
  const int ntile = n >> 4;
  Wsw[(((size_t)kt * (N >> 4) + ntile) * 32 + lane) * 16 + e] = (__bf16)W[i];
}

// ---------------------------------------------------------------------------
// WMMA GEMM: C[M,N] = relu?(A[M,K](lda) @ W[K,N] + bias[N]).
// grid = (M/16, N/64), block = 128 (4 waves, one 16x16 N-tile each).
// K compile-time -> fully unrolled v_wmma_f32_16x16x32_bf16 chain.
// A staged fp32->bf16 into fragment-order LDS via float2 + packed-pair
// stores (v_cvt_pk_bf16_f32 + ds_store_b32); fragments read back as
// 2x ds_load_b128.  B read as 2x global_load_b128 (wave-coalesced 1KB).
// ---------------------------------------------------------------------------
template <int K>
__global__ __launch_bounds__(128) void gemm_wmma_kernel(
    const float* __restrict__ A, int lda, const __bf16* __restrict__ Wsw,
    const float* __restrict__ bias, float* __restrict__ C, int N, int do_relu) {
  __shared__ __attribute__((aligned(32))) __bf16 sAf[512];  // [lane][e] fragment order

  const int tileM   = blockIdx.x * 16;
  const int wave    = threadIdx.x >> 5;
  const int lane    = threadIdx.x & 31;
  const int tileN16 = blockIdx.y * 4 + wave;   // 16-column N tile index
  const int ntiles  = N >> 4;

  v8f acc = {};
#pragma unroll
  for (int kt = 0; kt < K / 32; ++kt) {
    __syncthreads();
    // Stage A 16x32 tile: 256 adjacent-K pairs, 2 per thread.
    // Pair (r, kk=2*kp) and (r, kk+1) land in adjacent LDS bf16 slots.
#pragma unroll
    for (int p = threadIdx.x; p < 256; p += 128) {
      const int r = p >> 4, kp = p & 15, kk = kp << 1, g = kk >> 3;
      const int sl = r + ((g & 1) << 4);
      const int se = (kk & 7) + ((g >> 1) << 3);
      const float2 a2 = *(const float2*)&A[(size_t)(tileM + r) * lda + kt * 32 + kk];
      v2bf pk = { (__bf16)a2.x, (__bf16)a2.y };
      *(v2bf*)(sAf + sl * 16 + se) = pk;
    }
    __syncthreads();

    if (kt + 1 < K / 32)  // gfx1250 global_prefetch_b8 on next A K-tile
      __builtin_prefetch(&A[(size_t)(tileM + (threadIdx.x >> 3)) * lda + (kt + 1) * 32], 0, 3);

    v16bf a_frag = *(const v16bf*)(sAf + lane * 16);
    v16bf b_frag = *(const v16bf*)(Wsw + (((size_t)kt * ntiles + tileN16) * 32 + lane) * 16);

    acc = __builtin_amdgcn_wmma_f32_16x16x32_bf16(
        /*neg_a=*/false, a_frag, /*neg_b=*/false, b_frag,
        /*c_mod=*/(short)0, acc, /*reuse_a=*/false, /*reuse_b=*/false);
  }

  // D layout: element r -> M = tileM + r + 8*(lane/16), N = tileN16*16 + lane%16
  const int n     = tileN16 * 16 + (lane & 15);
  const int mbase = tileM + ((lane >> 4) << 3);
  const float bv  = bias[n];
#pragma unroll
  for (int r = 0; r < 8; ++r) {
    float out = acc[r] + bv;
    if (do_relu) out = fmaxf(out, 0.0f);
    C[(size_t)(mbase + r) * N + n] = out;
  }
}

// ---------------------------------------------------------------------------
// Neighbor aggregation: out[bv,c] = fo[bv,c] + max_n relu(dhat . sd[:,c]) *
// fo[nbr, 128+c].  One block per vertex, thread = channel.
// Phase 1: threads n<Kuse compute unit directions + cache nbr ids in LDS
// (100 sqrt per block instead of 12800).  Phase 2: per-channel max loop.
// ---------------------------------------------------------------------------
__global__ __launch_bounds__(128) void aggregate_kernel(
    const float* __restrict__ verts, const int* __restrict__ idx, int Kuse,
    const float* __restrict__ fo, const float* __restrict__ sd,
    float* __restrict__ out) {
  const int bv = blockIdx.x;
  const int b  = bv / V_;
  const int c  = threadIdx.x;

  __shared__ float sdx[KMAX_], sdy[KMAX_], sdz[KMAX_];
  __shared__ int   snb[KMAX_];

  const float px = verts[(size_t)bv * 3 + 0];
  const float py = verts[(size_t)bv * 3 + 1];
  const float pz = verts[(size_t)bv * 3 + 2];

  for (int n = threadIdx.x; n < Kuse; n += 128) {
    const int nb = idx[(size_t)bv * KMAX_ + n];
    const float* q = verts + ((size_t)b * V_ + nb) * 3;
    float dx = q[0] - px, dy = q[1] - py, dz = q[2] - pz;
    const float inv = 1.0f / fmaxf(sqrtf(dx * dx + dy * dy + dz * dz), 1e-12f);
    sdx[n] = dx * inv; sdy[n] = dy * inv; sdz[n] = dz * inv;
    snb[n] = nb;
  }
  __syncthreads();

  const float s0 = sd[c], s1 = sd[DIM_ + c], s2 = sd[2 * DIM_ + c];
  float acc = -__builtin_inff();
  for (int n = 0; n < Kuse; ++n) {
    const float theta = fmaxf(sdx[n] * s0 + sdy[n] * s1 + sdz[n] * s2, 0.0f);
    const float supp  = fo[((size_t)b * V_ + snb[n]) * 256 + DIM_ + c];
    acc = fmaxf(acc, theta * supp);
  }
  out[(size_t)bv * DIM_ + c] = fo[(size_t)bv * 256 + c] + acc;
}

// ---------------------------------------------------------------------------
// BN stats: one block per channel (128 blocks x 256 threads).
// ---------------------------------------------------------------------------
__global__ __launch_bounds__(256) void bn_stats_kernel(const float* __restrict__ x,
                                                       float* __restrict__ mu,
                                                       float* __restrict__ var) {
  const int c = blockIdx.x;
  __shared__ float s1[256], s2[256];
  float a = 0.f, q = 0.f;
  for (int m = threadIdx.x; m < ROWS_; m += 256) {
    float v = x[(size_t)m * DIM_ + c];
    a += v; q += v * v;
  }
  s1[threadIdx.x] = a; s2[threadIdx.x] = q;
  __syncthreads();
  for (int w = 128; w > 0; w >>= 1) {
    if (threadIdx.x < w) {
      s1[threadIdx.x] += s1[threadIdx.x + w];
      s2[threadIdx.x] += s2[threadIdx.x + w];
    }
    __syncthreads();
  }
  if (threadIdx.x == 0) {
    float m1 = s1[0] / (float)ROWS_;
    mu[c]  = m1;
    var[c] = s2[0] / (float)ROWS_ - m1 * m1;
  }
}

// BN + ReLU, writing into a strided slice (ldy) of the concat buffer.
__global__ void bn_apply_kernel(const float* __restrict__ x,
                                const float* __restrict__ mu, const float* __restrict__ var,
                                const float* __restrict__ gamma, const float* __restrict__ beta,
                                float* __restrict__ y, int ldy, int total) {
  const int i = blockIdx.x * blockDim.x + threadIdx.x;
  if (i >= total) return;
  const int m = i >> 7;          // row
  const int c = i & (DIM_ - 1);  // channel
  float v = (x[i] - mu[c]) * rsqrtf(var[c] + 1e-5f) * gamma[c] + beta[c];
  y[(size_t)m * ldy + c] = fmaxf(v, 0.0f);
}

// ---------------------------------------------------------------------------
extern "C" void kernel_launch(void* const* d_in, const int* in_sizes, int n_in,
                              void* d_out, int out_size, void* d_ws, size_t ws_size,
                              hipStream_t stream) {
  const float* vertices  = (const float*)d_in[0];   // (2,1024,3)
  const float* x_in      = (const float*)d_in[1];   // (2,1024,128)
  const float* conv_W    = (const float*)d_in[2];   // (6,128,256)
  const float* conv_b    = (const float*)d_in[3];   // (6,256)
  const float* conv_dirs = (const float*)d_in[4];   // (6,3,128)
  const float* bn_gamma  = (const float*)d_in[5];   // (6,128)
  const float* bn_beta   = (const float*)d_in[6];   // (6,128)
  const float* down_W    = (const float*)d_in[7];   // (384,256)
  const float* down_b    = (const float*)d_in[8];   // (256,)
  float* out = (float*)d_out;                       // (2,1024,256)

  // Workspace carve-out (256B aligned slices).
  char*  ws  = (char*)d_ws;
  size_t cur = 0;
  auto alloc = [&](size_t bytes) -> char* {
    char* p = ws + cur;
    cur += (bytes + 255) & ~(size_t)255;
    return p;
  };
  int*    idx = (int*)   alloc((size_t)ROWS_ * KMAX_ * 4);  // 800 KB
  float*  fo  = (float*) alloc((size_t)ROWS_ * 256 * 4);    // 2 MB
  float*  tmp = (float*) alloc((size_t)ROWS_ * DIM_ * 4);   // 1 MB
  float*  cat = (float*) alloc((size_t)ROWS_ * 384 * 4);    // 3 MB (fl|fm|fg slices)
  __bf16* Wsw = (__bf16*)alloc((size_t)384 * 256 * 2);      // 192 KB (max K=384)
  float*  sd  = (float*) alloc(3 * DIM_ * 4);
  float*  mu  = (float*) alloc(DIM_ * 4);
  float*  var = (float*) alloc(DIM_ * 4);
  (void)ws_size; (void)in_sizes; (void)n_in; (void)out_size;

  // 1) kNN once; k=5/20/100 are prefixes of the sorted neighbor list.
  knn_kernel<<<ROWS_, 128, 0, stream>>>(vertices, idx);

  const dim3 ggrid(ROWS_ / 16, 256 / 64);

  // Per-layer fused conv + BN-ReLU; output goes into a 128-wide slice of cat.
  auto conv_bn = [&](const float* fin, int lda, int li, int Kuse, float* fout) {
    sd_kernel<<<1, 128, 0, stream>>>(conv_dirs + (size_t)li * 3 * DIM_, sd);
    wswizzle_kernel<<<(DIM_ * 256 + 255) / 256, 256, 0, stream>>>(
        conv_W + (size_t)li * DIM_ * 256, Wsw, DIM_, 256, DIM_ * 256);
    gemm_wmma_kernel<DIM_><<<ggrid, 128, 0, stream>>>(
        fin, lda, Wsw, conv_b + (size_t)li * 256, fo, 256, 0);
    aggregate_kernel<<<ROWS_, 128, 0, stream>>>(vertices, idx, Kuse, fo, sd, tmp);
    bn_stats_kernel<<<DIM_, 256, 0, stream>>>(tmp, mu, var);
    bn_apply_kernel<<<(ROWS_ * DIM_ + 255) / 256, 256, 0, stream>>>(
        tmp, mu, var, bn_gamma + (size_t)li * DIM_, bn_beta + (size_t)li * DIM_,
        fout, 384, ROWS_ * DIM_);
  };

  float* fl = cat + 0;    // (2048,384) slices
  float* fm = cat + 128;
  float* fg = cat + 256;

  conv_bn(x_in, 128, 0, 5,   fl);
  conv_bn(x_in, 128, 1, 20,  fm);
  conv_bn(fm,   384, 2, 20,  fm);
  conv_bn(x_in, 128, 3, 100, fg);
  conv_bn(fg,   384, 4, 100, fg);
  conv_bn(fg,   384, 3, 100, fg);   // re-uses W[3]/b[3]/dirs[3]/gamma[3]/beta[3]

  // Final down-projection: WMMA GEMM (K=384) with fused bias+ReLU into d_out.
  wswizzle_kernel<<<(384 * 256 + 255) / 256, 256, 0, stream>>>(
      down_W, Wsw, 384, 256, 384 * 256);
  gemm_wmma_kernel<384><<<ggrid, 128, 0, stream>>>(
      cat, 384, Wsw, down_b, out, 256, 1);
}
